// TTTAttention_53025666236920
// MI455X (gfx1250) — matrix operand order repaired and checked by reading, not compiled, not run on hardware
//
#include <hip/hip_runtime.h>

// ---------------------------------------------------------------------------
// CDNA5 (gfx1250) TTT attention. wave32, WMMA bf16 16x16x32, f32 accumulate.
// B-side operands stored transposed (n-major) -> every WMMA fragment is two
// contiguous 16B LDS loads per lane. Global->LDS staging uses the CDNA5
// async-to-LDS path (ASYNCcnt) with double buffering where available.
// ---------------------------------------------------------------------------

typedef __attribute__((ext_vector_type(16))) __bf16 v16bf;
typedef __attribute__((ext_vector_type(8)))  __bf16 v8bf;
typedef __attribute__((ext_vector_type(8)))  float  v8f;

__device__ __forceinline__ v8f wmma_bf16(v16bf a, v16bf b, v8f c) {
  // (neg_a, A, neg_b, B, c_mod, C, reuse_a, reuse_b) -> v_wmma_f32_16x16x32_bf16
  return __builtin_amdgcn_wmma_f32_16x16x32_bf16(false, a, false, b, (short)0, c,
                                                 false, false);
}

// 16-byte global -> LDS copy. Uses GLOBAL_LOAD_ASYNC_TO_LDS_B128 (ASYNCcnt,
// no VGPR data) when the toolchain exposes it; else a plain wide copy.
// Builtin signature: (addrspace(1) int4*, addrspace(3) int4*, imm off, imm cpol).
__device__ __forceinline__ void async_copy16(const __bf16* g, __bf16* l) {
#if defined(__AMDGCN__) && __has_builtin(__builtin_amdgcn_global_load_async_to_lds_b128)
  typedef int v4i __attribute__((vector_size(16)));
  typedef __attribute__((address_space(1))) v4i gv4i;
  typedef __attribute__((address_space(3))) v4i lv4i;
  __builtin_amdgcn_global_load_async_to_lds_b128((gv4i*)g, (lv4i*)l, 0, 0);
#else
  *(v8bf*)l = *(const v8bf*)g;
#endif
}

__device__ __forceinline__ void async_wait_all() {
#if defined(__AMDGCN__) && __has_builtin(__builtin_amdgcn_s_wait_asynccnt)
  __builtin_amdgcn_s_wait_asynccnt(0);
#endif
}

// A-fragment from a row-major 16xK LDS tile. ISA 7.12.2 16-bit A 16x32:
// lane L: row M=L%16; low lanes K={0..7,16..23}, high lanes K={8..15,24..31}
// -> two contiguous 8-element (16B) runs per lane.
__device__ __forceinline__ v16bf frag_a(const __bf16* p, int ld, int row0, int k0,
                                        int lane) {
  const int r  = lane & 15;
  const int kh = (lane & 16) ? 8 : 0;
  const __bf16* q = p + (row0 + r) * ld + k0 + kh;
  const v8bf lo = *(const v8bf*)q;
  const v8bf hi = *(const v8bf*)(q + 16);
  return __builtin_shufflevector(lo, hi, 0, 1, 2, 3, 4, 5, 6, 7, 8, 9, 10, 11, 12,
                                 13, 14, 15);
}

// B-fragment (logical B[k][n]) from TRANSPOSED storage t[n][k]: low lanes
// K={0..7,16..23}, high lanes K={8..15,24..31} -> two 16B runs per lane.
__device__ __forceinline__ v16bf frag_bt(const __bf16* p, int ld, int k0, int col0,
                                         int lane) {
  const int c  = lane & 15;
  const int kh = (lane & 16) ? 8 : 0;
  const __bf16* q = p + (col0 + c) * ld + k0 + kh;
  const v8bf lo = *(const v8bf*)q;
  const v8bf hi = *(const v8bf*)(q + 16);
  return __builtin_shufflevector(lo, hi, 0, 1, 2, 3, 4, 5, 6, 7, 8, 9, 10, 11, 12,
                                 13, 14, 15);
}

// ---------------------------------------------------------------------------
// Tiled GEMM: C[M,N] = A[M,K](bf16, row-major) * B (given as Bt[N,K] bf16)
// (+ bias). 64x64 tile / 128 threads (4 wave32), double-buffered async staging.
// ---------------------------------------------------------------------------
__global__ __launch_bounds__(128) void gemm_bf16_kernel(
    const __bf16* __restrict__ A, const __bf16* __restrict__ Bt,
    const float* __restrict__ bias, float* __restrict__ C, int M, int N, int K) {
  __shared__ __attribute__((aligned(16))) __bf16 As[2][64 * 32];   // [m][k]
  __shared__ __attribute__((aligned(16))) __bf16 Bst[2][64 * 32];  // [n][k]
  const int tid = threadIdx.x, lane = tid & 31, wave = tid >> 5;
  const int m0 = blockIdx.y * 64, n0 = blockIdx.x * 64;
  const v8f vzero = {0.f, 0.f, 0.f, 0.f, 0.f, 0.f, 0.f, 0.f};
  v8f acc[4];
#pragma unroll
  for (int t = 0; t < 4; ++t) acc[t] = vzero;

  const int sr = tid >> 1, sc = (tid & 1) * 16;  // 64 rows x (2 x 16) staging

  auto stage = [&](int buf, int k0) {
    const __bf16* ga = A + (long)(m0 + sr) * K + k0 + sc;
    const __bf16* gb = Bt + (long)(n0 + sr) * K + k0 + sc;
    async_copy16(ga,     &As[buf][sr * 32 + sc]);
    async_copy16(ga + 8, &As[buf][sr * 32 + sc + 8]);
    async_copy16(gb,     &Bst[buf][sr * 32 + sc]);
    async_copy16(gb + 8, &Bst[buf][sr * 32 + sc + 8]);
  };

  stage(0, 0);
  async_wait_all();
  __syncthreads();

  int cur = 0;
  for (int k0 = 0; k0 < K; k0 += 32) {
    if (k0 + 32 < K) {
      stage(cur ^ 1, k0 + 32);  // DMA next slab while WMMAs run on this one
      __builtin_prefetch(A + (long)(m0 + sr) * K + k0 + 64 + sc, 0, 3);
      __builtin_prefetch(Bt + (long)(n0 + sr) * K + k0 + 64 + sc, 0, 3);
    }
    const v16bf af = frag_a(As[cur], 32, wave * 16, 0, lane);
#pragma unroll
    for (int nt = 0; nt < 4; ++nt) {
      const v16bf bf = frag_bt(Bst[cur], 32, 0, nt * 16, lane);
      acc[nt] = wmma_bf16(af, bf, acc[nt]);
    }
    async_wait_all();
    __syncthreads();
    cur ^= 1;
  }
  const int nn = lane & 15, mh = (lane & 16) ? 8 : 0;
#pragma unroll
  for (int nt = 0; nt < 4; ++nt) {
    const int col = n0 + nt * 16 + nn;
    const float bv = bias ? bias[col] : 0.f;
#pragma unroll
    for (int r = 0; r < 8; ++r)
      C[(long)(m0 + wave * 16 + mh + r) * N + col] = acc[nt][r] + bv;
  }
}

// ---------------------------------------------------------------------------
__global__ void f32_to_bf16_kernel(const float* __restrict__ in,
                                   __bf16* __restrict__ out, long n) {
  const long i = (long)blockIdx.x * blockDim.x + threadIdx.x;
  if (i < n) out[i] = (__bf16)in[i];
}

// out[c*R + r] = (bf16) in[r*C + c]   (weights -> transposed bf16)
__global__ void f32_to_bf16_t_kernel(const float* __restrict__ in,
                                     __bf16* __restrict__ out, int R, int C) {
  const long i = (long)blockIdx.x * blockDim.x + threadIdx.x;
  if (i >= (long)R * C) return;
  const int r = (int)(i / C), c = (int)(i % C);
  out[(long)c * R + r] = (__bf16)in[i];
}

// lr[b,m,i,h] = sigmoid(xx[b, m*16+i, :] @ lr_w[:,h] + lr_b[h]) / 64
__global__ void lr_kernel(const float* __restrict__ x, const float* __restrict__ lr_w,
                          const float* __restrict__ lr_b, float* __restrict__ lrout) {
  const long e = (long)blockIdx.x * blockDim.x + threadIdx.x;
  if (e >= (long)16 * 64 * 16 * 24) return;
  int h = (int)(e % 24);
  long r = e / 24;
  const int i = (int)(r % 16); r /= 16;
  const int m = (int)(r % 64);
  const int b = (int)(r / 64);
  const int n = m * 16 + i;
  const float* xr = x + ((long)b * 1024 + n) * 768;
  const float* xf = x + ((long)b * 1024 + (1023 - n)) * 768;
  float acc = lr_b[h];
  for (int c = 0; c < 768; ++c) acc += xr[c] * lr_w[c * 24 + h];
  for (int c = 0; c < 768; ++c) acc += xf[c] * lr_w[(768 + c) * 24 + h];
  const float s = 1.f / (1.f + expf(-acc));
  lrout[e] = s * (1.0f / 64.0f);
}

// Head split (seq-flip for heads 12..23) + rotary (pos = n % 16); f32+bf16 out.
__global__ void prepare_heads_kernel(const float* __restrict__ qkv,
                                     float* __restrict__ XQf, float* __restrict__ XKf,
                                     float* __restrict__ XVf,
                                     __bf16* __restrict__ XQb,
                                     __bf16* __restrict__ XKb) {
  const long e = (long)blockIdx.x * blockDim.x + threadIdx.x;
  if (e >= (long)16 * 24 * 1024 * 32) return;
  const int t = (int)(e & 31);
  long r = e >> 5;
  const int n = (int)(r & 1023); r >>= 10;
  const int h = (int)(r % 24);
  const int b = (int)(r / 24);
  const int d0 = 2 * t;
  const int hs = (h < 12) ? h : (h - 12);
  const int sn = (h < 12) ? n : (1023 - n);
  const float* p = qkv + ((long)(b * 1024 + sn)) * 2304 + hs * 64 + d0;
  const float q0 = p[0], q1 = p[1];
  const float k0 = p[768], k1 = p[769];
  const float v0 = p[1536], v1 = p[1537];
  const float pos = (float)(n & 15);
  const float ang = pos * powf(10000.f, -(float)d0 * (1.f / 64.f));
  const float cs = cosf(ang), si = sinf(ang);
  const float Q0 = q0 * cs - q1 * si, Q1 = q0 * si + q1 * cs;
  const float K0 = k0 * cs - k1 * si, K1 = k0 * si + k1 * cs;
  const long o = (((long)(b * 24 + h)) * 1024 + n) * 64 + d0;
  XQf[o] = Q0; XQf[o + 1] = Q1;
  XKf[o] = K0; XKf[o + 1] = K1;
  XVf[o] = v0; XVf[o + 1] = v1;
  XQb[o] = (__bf16)Q0; XQb[o + 1] = (__bf16)Q1;
  XKb[o] = (__bf16)K0; XKb[o + 1] = (__bf16)K1;
}

// ---------------------------------------------------------------------------
// TTT scan: one 128-thread workgroup (4 wave32) per (batch, head).
// W in LDS: f32 master Wf[64][65] + bf16 transposed mirror WbT[n][k].
// ---------------------------------------------------------------------------
__global__ __launch_bounds__(128) void ttt_scan_kernel(
    const __bf16* __restrict__ XQb, const __bf16* __restrict__ XKb,
    const float* __restrict__ XQf, const float* __restrict__ XKf,
    const float* __restrict__ XVf, const float* __restrict__ lrA,
    const float* __restrict__ W1, const float* __restrict__ b1,
    const float* __restrict__ ln_g, const float* __restrict__ ln_b,
    const float* __restrict__ tok_delta, float* __restrict__ Zout) {
  const int bh = blockIdx.x;
  const int h = bh % 24, b = bh / 24;
  const int tid = threadIdx.x, lane = tid & 31, wave = tid >> 5;
  const v8f vzero = {0.f, 0.f, 0.f, 0.f, 0.f, 0.f, 0.f, 0.f};

  __shared__ __attribute__((aligned(16))) float  Wf[64 * 65];
  __shared__ __attribute__((aligned(16))) __bf16 WbT[64 * 64];   // [n][k]
  __shared__ __attribute__((aligned(16))) float  bbv[64], gln[64], blnv[64];
  __shared__ __attribute__((aligned(16))) __bf16 xqb[16 * 64], xkb[16 * 64];
  __shared__ __attribute__((aligned(16))) float  xqf[16 * 64], xkf[16 * 64],
      xvf[16 * 64];
  __shared__ __attribute__((aligned(16))) float  Zt[16 * 64];
  __shared__ __attribute__((aligned(16))) __bf16 gradT[64 * 32];  // [n][k], k>=16 zero
  __shared__ __attribute__((aligned(16))) float  attn[16 * 16];
  __shared__ __attribute__((aligned(16))) __bf16 Sb[16 * 32], Tb[16 * 32];
  __shared__ __attribute__((aligned(16))) __bf16 Au[64 * 32];     // [d][k], k>=16 zero
  __shared__ float lrv[16], tokv[16], lastv[16];
  __shared__ float red1[16 * 8], red2[16 * 8];
  __shared__ float mu_s[16], rs_s[16], s1_s[16], s2_s[16];

  for (int i = tid; i < 64 * 64; i += 128) {
    const float w = W1[(long)h * 4096 + i];
    Wf[(i >> 6) * 65 + (i & 63)] = w;
    WbT[(i & 63) * 64 + (i >> 6)] = (__bf16)w;  // transpose: [n][k]
  }
  if (tid < 64) {
    bbv[tid]  = b1[h * 64 + tid];
    gln[tid]  = ln_g[h * 64 + tid];
    blnv[tid] = ln_b[h * 64 + tid];
  }
  for (int i = tid; i < 64 * 32; i += 128) { gradT[i] = (__bf16)0.f; Au[i] = (__bf16)0.f; }
  if (tid < 16) tokv[tid] = fmaxf(1.f / (float)(tid + 1) + tok_delta[tid], 0.f);
  __syncthreads();

  const long base = (long)bh * 1024 * 64;

  for (int m = 0; m < 64; ++m) {
    // ---- stage minibatch tiles: bf16 via async DMA, f32 via float4 ----
    {
      const int i0 = tid * 8;
      const long g = base + (long)m * 16 * 64 + i0;
      async_copy16(XQb + g, xqb + i0);   // 8 bf16 = 16B per thread
      async_copy16(XKb + g, xkb + i0);
      const float4 q0 = *(const float4*)(XQf + g);
      const float4 q1 = *(const float4*)(XQf + g + 4);
      const float4 k0 = *(const float4*)(XKf + g);
      const float4 k1 = *(const float4*)(XKf + g + 4);
      const float4 v0 = *(const float4*)(XVf + g);
      const float4 v1 = *(const float4*)(XVf + g + 4);
      *(float4*)(xqf + i0) = q0; *(float4*)(xqf + i0 + 4) = q1;
      *(float4*)(xkf + i0) = k0; *(float4*)(xkf + i0 + 4) = k1;
      *(float4*)(xvf + i0) = v0; *(float4*)(xvf + i0 + 4) = v1;
    }
    if (tid < 16) {
      lrv[tid] = lrA[((long)((b * 64 + m) * 16 + tid)) * 24 + h];
      lastv[tid] = tokv[15] * lrv[tid];
    }
    async_wait_all();
    __syncthreads();

    // ---- Z1 = xk @ W + bb  (2 WMMAs per wave) ----
    {
      v8f acc = vzero;
      const v16bf a0 = frag_a(xkb, 64, 0, 0, lane);
      const v16bf a1 = frag_a(xkb, 64, 0, 32, lane);
      const v16bf w0 = frag_bt(WbT, 64, 0, wave * 16, lane);
      const v16bf w1 = frag_bt(WbT, 64, 32, wave * 16, lane);
      acc = wmma_bf16(a0, w0, acc);
      acc = wmma_bf16(a1, w1, acc);
      const int nn = lane & 15, mh = (lane & 16) ? 8 : 0;
#pragma unroll
      for (int r = 0; r < 8; ++r)
        Zt[(mh + r) * 64 + wave * 16 + nn] = acc[r] + bbv[wave * 16 + nn];
    }
    __syncthreads();

    // ---- grad = ln_l2_bwd(Z1, xv - xk, g, b) ----
    {
      const int row = tid >> 3, sub = tid & 7;
      float z[8], tg[8];
      {
        const float4 z0 = *(const float4*)(Zt + row * 64 + sub * 8);
        const float4 z1 = *(const float4*)(Zt + row * 64 + sub * 8 + 4);
        const float4 va = *(const float4*)(xvf + row * 64 + sub * 8);
        const float4 vb = *(const float4*)(xvf + row * 64 + sub * 8 + 4);
        const float4 ka = *(const float4*)(xkf + row * 64 + sub * 8);
        const float4 kb = *(const float4*)(xkf + row * 64 + sub * 8 + 4);
        z[0] = z0.x; z[1] = z0.y; z[2] = z0.z; z[3] = z0.w;
        z[4] = z1.x; z[5] = z1.y; z[6] = z1.z; z[7] = z1.w;
        tg[0] = va.x - ka.x; tg[1] = va.y - ka.y; tg[2] = va.z - ka.z; tg[3] = va.w - ka.w;
        tg[4] = vb.x - kb.x; tg[5] = vb.y - kb.y; tg[6] = vb.z - kb.z; tg[7] = vb.w - kb.w;
      }
      float s = 0.f, s2 = 0.f;
#pragma unroll
      for (int j = 0; j < 8; ++j) { s += z[j]; s2 += z[j] * z[j]; }
      red1[row * 8 + sub] = s; red2[row * 8 + sub] = s2;
      __syncthreads();
      if (sub == 0) {
        float a = 0.f, c = 0.f;
#pragma unroll
        for (int j = 0; j < 8; ++j) { a += red1[row * 8 + j]; c += red2[row * 8 + j]; }
        const float mu = a * (1.f / 64.f);
        mu_s[row] = mu;
        rs_s[row] = rsqrtf(c * (1.f / 64.f) - mu * mu + 1e-6f);
      }
      __syncthreads();
      const float mu = mu_s[row], rs = rs_s[row];
      float p1 = 0.f, p2 = 0.f;
      float gxh[8], xh[8];
#pragma unroll
      for (int j = 0; j < 8; ++j) {
        const int c = sub * 8 + j;
        xh[j] = (z[j] - mu) * rs;
        const float gout = gln[c] * xh[j] + blnv[c] - tg[j];
        gxh[j] = gout * gln[c];
        p1 += gxh[j]; p2 += gxh[j] * xh[j];
      }
      red1[row * 8 + sub] = p1; red2[row * 8 + sub] = p2;
      __syncthreads();
      if (sub == 0) {
        float a = 0.f, c = 0.f;
#pragma unroll
        for (int j = 0; j < 8; ++j) { a += red1[row * 8 + j]; c += red2[row * 8 + j]; }
        s1_s[row] = a; s2_s[row] = c;
      }
      __syncthreads();
      const float S1 = s1_s[row], S2 = s2_s[row];
#pragma unroll
      for (int j = 0; j < 8; ++j) {
        const int c = sub * 8 + j;
        const float gr = (64.f * gxh[j] - S1 - xh[j] * S2) * (rs * (1.f / 64.f));
        gradT[c * 32 + row] = (__bf16)gr;  // transposed store [n][k]
      }
    }
    __syncthreads();

    // ---- attn = xq @ xk^T  (wave 0; xkb read as transposed-B, contiguous) ----
    if (wave == 0) {
      v8f acc = vzero;
      const v16bf a0 = frag_a(xqb, 64, 0, 0, lane);
      const v16bf a1 = frag_a(xqb, 64, 0, 32, lane);
      const v16bf t0 = frag_bt(xkb, 64, 0, 0, lane);
      const v16bf t1 = frag_bt(xkb, 64, 32, 0, lane);
      acc = wmma_bf16(a0, t0, acc);
      acc = wmma_bf16(a1, t1, acc);
      const int nn = lane & 15, mh = (lane & 16) ? 8 : 0;
#pragma unroll
      for (int r = 0; r < 8; ++r) attn[(mh + r) * 16 + nn] = acc[r];
    }
    __syncthreads();

    // ---- Sb = -(eta*mask*attn), Tb = -(eta*mask); K-padded to 32 ----
    for (int i = tid; i < 16 * 32; i += 128) {
      const int rr = i >> 5, cc = i & 31;
      const float e = (cc < 16 && cc <= rr) ? tokv[rr] * lrv[cc] : 0.f;
      const float av = (cc < 16) ? attn[rr * 16 + cc] : 0.f;
      Sb[i] = (__bf16)(-(e * av));
      Tb[i] = (__bf16)(-e);
    }
    __syncthreads();

    // ---- Z1_bar = xq@W + Sb@grad + Tb@grad + bb  (4 WMMAs per wave) ----
    {
      v8f acc = vzero;
      const v16bf a0 = frag_a(xqb, 64, 0, 0, lane);
      const v16bf a1 = frag_a(xqb, 64, 0, 32, lane);
      const v16bf w0 = frag_bt(WbT, 64, 0, wave * 16, lane);
      const v16bf w1 = frag_bt(WbT, 64, 32, wave * 16, lane);
      acc = wmma_bf16(a0, w0, acc);
      acc = wmma_bf16(a1, w1, acc);
      const v16bf sA = frag_a(Sb, 32, 0, 0, lane);
      const v16bf tA = frag_a(Tb, 32, 0, 0, lane);
      const v16bf gB = frag_bt(gradT, 32, 0, wave * 16, lane);
      acc = wmma_bf16(sA, gB, acc);
      acc = wmma_bf16(tA, gB, acc);
      const int nn = lane & 15, mh = (lane & 16) ? 8 : 0;
#pragma unroll
      for (int r = 0; r < 8; ++r)
        Zt[(mh + r) * 64 + wave * 16 + nn] = acc[r] + bbv[wave * 16 + nn];
    }
    __syncthreads();

    // ---- out = xq + LN(Z1_bar); write Zout[bh][n][d] (float4) ----
    {
      const int row = tid >> 3, sub = tid & 7;
      float z[8];
      {
        const float4 z0 = *(const float4*)(Zt + row * 64 + sub * 8);
        const float4 z1 = *(const float4*)(Zt + row * 64 + sub * 8 + 4);
        z[0] = z0.x; z[1] = z0.y; z[2] = z0.z; z[3] = z0.w;
        z[4] = z1.x; z[5] = z1.y; z[6] = z1.z; z[7] = z1.w;
      }
      float s = 0.f, s2 = 0.f;
#pragma unroll
      for (int j = 0; j < 8; ++j) { s += z[j]; s2 += z[j] * z[j]; }
      red1[row * 8 + sub] = s; red2[row * 8 + sub] = s2;
      __syncthreads();
      if (sub == 0) {
        float a = 0.f, c = 0.f;
#pragma unroll
        for (int j = 0; j < 8; ++j) { a += red1[row * 8 + j]; c += red2[row * 8 + j]; }
        const float mu = a * (1.f / 64.f);
        mu_s[row] = mu;
        rs_s[row] = rsqrtf(c * (1.f / 64.f) - mu * mu + 1e-6f);
      }
      __syncthreads();
      const float mu = mu_s[row], rs = rs_s[row];
      const float4 qa = *(const float4*)(xqf + row * 64 + sub * 8);
      const float4 qb2 = *(const float4*)(xqf + row * 64 + sub * 8 + 4);
      float o[8];
      const float qv[8] = {qa.x, qa.y, qa.z, qa.w, qb2.x, qb2.y, qb2.z, qb2.w};
#pragma unroll
      for (int j = 0; j < 8; ++j) {
        const int c = sub * 8 + j;
        o[j] = qv[j] + gln[c] * ((z[j] - mu) * rs) + blnv[c];
      }
      float* dst = Zout + base + (long)(m * 16 + row) * 64 + sub * 8;
      *(float4*)dst       = make_float4(o[0], o[1], o[2], o[3]);
      *(float4*)(dst + 4) = make_float4(o[4], o[5], o[6], o[7]);
    }
    __syncthreads();

    // ---- Au[d][k] = -last[k]*xk[k][d] (k<16; pad zero), bb update ----
    {
      const int d = tid >> 1, kb = (tid & 1) * 8;
#pragma unroll
      for (int j = 0; j < 8; ++j) {
        const int k = kb + j;
        Au[d * 32 + k] = (__bf16)(-(lastv[k] * xkf[k * 64 + d]));
      }
    }
    if (tid < 64) {
      float acc2 = bbv[tid];
#pragma unroll
      for (int k2 = 0; k2 < 16; ++k2)
        acc2 -= lastv[k2] * (float)gradT[tid * 32 + k2];
      bbv[tid] = acc2;
    }
    __syncthreads();

    // ---- W += Au @ grad (i.e. W -= (last*xk)^T @ grad); 4 tiles per wave ----
    {
#pragma unroll
      for (int i2 = 0; i2 < 4; ++i2) {
        const int t = wave * 4 + i2;
        const int mt = t >> 2, nt = t & 3;
        const v16bf aA = frag_a(Au, 32, mt * 16, 0, lane);
        const v16bf gB = frag_bt(gradT, 32, 0, nt * 16, lane);
        const int nn = lane & 15, mh = (lane & 16) ? 8 : 0;
        v8f cf;
#pragma unroll
        for (int r2 = 0; r2 < 8; ++r2)
          cf[r2] = Wf[(mt * 16 + mh + r2) * 65 + nt * 16 + nn];
        const v8f d = wmma_bf16(aA, gB, cf);
        v8bf wt;
#pragma unroll
        for (int r2 = 0; r2 < 8; ++r2) {
          Wf[(mt * 16 + mh + r2) * 65 + nt * 16 + nn] = d[r2];
          wt[r2] = (__bf16)d[r2];
        }
        // WbT[n][k]: fixed n = nt*16+nn, k = mt*16+mh .. +8 contiguous
        *(v8bf*)(WbT + (nt * 16 + nn) * 64 + mt * 16 + mh) = wt;
      }
    }
    __syncthreads();
  }
}

// ---------------------------------------------------------------------------
// Post-LN over W2=1536 per (b, n); gathers head-major Z on the fly.
// ---------------------------------------------------------------------------
__global__ __launch_bounds__(256) void post_ln_kernel(const float* __restrict__ Zhead,
                                                      const float* __restrict__ g,
                                                      const float* __restrict__ bias,
                                                      float* __restrict__ Zn) {
  const int bn = blockIdx.x;
  const int n = bn & 1023, b = bn >> 10;
  __shared__ float red[256];
  __shared__ float stat[2];
  float vals[6];
  float s = 0.f, s2 = 0.f;
#pragma unroll
  for (int j = 0; j < 6; ++j) {
    const int c = threadIdx.x + 256 * j;
    const int hh = c >> 6, d = c & 63;
    const float v = Zhead[(((long)(b * 24 + hh)) * 1024 + n) * 64 + d];
    vals[j] = v; s += v; s2 += v * v;
  }
  red[threadIdx.x] = s;
  __syncthreads();
  if (threadIdx.x == 0) {
    float t = 0.f;
    for (int i = 0; i < 256; ++i) t += red[i];
    stat[0] = t * (1.f / 1536.f);
  }
  __syncthreads();
  red[threadIdx.x] = s2;
  __syncthreads();
  if (threadIdx.x == 0) {
    float t = 0.f;
    for (int i = 0; i < 256; ++i) t += red[i];
    const float mu = stat[0];
    stat[1] = rsqrtf(t * (1.f / 1536.f) - mu * mu + 1e-6f);
  }
  __syncthreads();
  const float mu = stat[0], rs = stat[1];
#pragma unroll
  for (int j = 0; j < 6; ++j) {
    const int c = threadIdx.x + 256 * j;
    Zn[(long)bn * 1536 + c] = g[c] * ((vals[j] - mu) * rs) + bias[c];
  }
}

// gated[b,n,c] = Zn[b,n,c] * Zn[b,1023-n,768+c]  (bf16 out, feeds final GEMM)
__global__ void gate_kernel(const float* __restrict__ Zn, __bf16* __restrict__ gated) {
  const long e = (long)blockIdx.x * blockDim.x + threadIdx.x;
  if (e >= (long)16 * 1024 * 768) return;
  const int c = (int)(e % 768);
  long r = e / 768;
  const int n = (int)(r % 1024);
  const int b = (int)(r / 1024);
  const float zf = Zn[((long)(b * 1024 + n)) * 1536 + c];
  const float zb = Zn[((long)(b * 1024 + (1023 - n))) * 1536 + 768 + c];
  gated[e] = (__bf16)(zf * zb);
}

// ---------------------------------------------------------------------------
extern "C" void kernel_launch(void* const* d_in, const int* in_sizes, int n_in,
                              void* d_out, int out_size, void* d_ws, size_t ws_size,
                              hipStream_t stream) {
  (void)in_sizes; (void)n_in; (void)out_size; (void)ws_size;
  const float* x         = (const float*)d_in[0];
  const float* qkv_w     = (const float*)d_in[1];
  const float* proj_w    = (const float*)d_in[2];
  const float* proj_b    = (const float*)d_in[3];
  const float* W1        = (const float*)d_in[4];
  const float* b1        = (const float*)d_in[5];
  const float* ttt_ln_w  = (const float*)d_in[6];
  const float* ttt_ln_b  = (const float*)d_in[7];
  const float* post_ln_w = (const float*)d_in[8];
  const float* post_ln_b = (const float*)d_in[9];
  const float* lr_w      = (const float*)d_in[10];
  const float* lr_b      = (const float*)d_in[11];
  const float* tokd      = (const float*)d_in[12];

  const long NE = (long)16 * 24 * 1024 * 64;  // head-elements per tensor
  char* w = (char*)d_ws;
  auto take = [&](size_t bytes) -> char* {
    char* p = w;
    w += (bytes + 255) & ~(size_t)255;
    return p;
  };
  __bf16* xb      = (__bf16*)take((size_t)16 * 1024 * 768 * 2);
  __bf16* qkvwT   = (__bf16*)take((size_t)768 * 2304 * 2);  // [2304][768]
  __bf16* projwT  = (__bf16*)take((size_t)768 * 768 * 2);   // [768][768]
  float*  qkvf    = (float*)take((size_t)16 * 1024 * 2304 * 4);
  float*  XQf     = (float*)take((size_t)NE * 4);
  float*  XKf     = (float*)take((size_t)NE * 4);
  float*  XVf     = (float*)take((size_t)NE * 4);
  __bf16* XQb     = (__bf16*)take((size_t)NE * 2);
  __bf16* XKb     = (__bf16*)take((size_t)NE * 2);
  float*  lrA     = (float*)take((size_t)16 * 64 * 16 * 24 * 4);
  float*  Zhead   = (float*)take((size_t)NE * 4);
  float*  Zn      = qkvf;  // qkv buffer is dead after prepare_heads
  __bf16* gated   = xb;    // x-bf16 buffer is dead after the QKV GEMM

  // 1) bf16 conversions (weights transposed for the B-side of the GEMMs)
  {
    long n = (long)16 * 1024 * 768;
    f32_to_bf16_kernel<<<(int)((n + 255) / 256), 256, 0, stream>>>(x, xb, n);
  }
  {
    long n = (long)768 * 2304;
    f32_to_bf16_t_kernel<<<(int)((n + 255) / 256), 256, 0, stream>>>(qkv_w, qkvwT,
                                                                     768, 2304);
  }
  {
    long n = (long)768 * 768;
    f32_to_bf16_t_kernel<<<(int)((n + 255) / 256), 256, 0, stream>>>(proj_w, projwT,
                                                                     768, 768);
  }

  // 2) qkv = x @ qkv_w   (WMMA bf16, async double-buffered staging)
  gemm_bf16_kernel<<<dim3(2304 / 64, 16384 / 64), 128, 0, stream>>>(
      xb, qkvwT, nullptr, qkvf, 16384, 2304, 768);

  // 3) learning rates
  {
    long n = (long)16 * 64 * 16 * 24;
    lr_kernel<<<(int)((n + 127) / 128), 128, 0, stream>>>(x, lr_w, lr_b, lrA);
  }

  // 4) head split + rotary
  {
    long n = (long)16 * 24 * 1024 * 32;
    prepare_heads_kernel<<<(int)((n + 255) / 256), 256, 0, stream>>>(
        qkvf, XQf, XKf, XVf, XQb, XKb);
  }

  // 5) sequential TTT scan: one workgroup per (b, h)
  ttt_scan_kernel<<<384, 128, 0, stream>>>(XQb, XKb, XQf, XKf, XVf, lrA, W1, b1,
                                           ttt_ln_w, ttt_ln_b, tokd, Zhead);

  // 6) post layer-norm over W2=1536
  post_ln_kernel<<<16 * 1024, 256, 0, stream>>>(Zhead, post_ln_w, post_ln_b, Zn);

  // 7) flip-gate -> bf16
  {
    long n = (long)16 * 1024 * 768;
    gate_kernel<<<(int)((n + 255) / 256), 256, 0, stream>>>(Zn, gated);
  }

  // 8) out = gated @ proj_w + proj_b   (WMMA bf16)
  gemm_bf16_kernel<<<dim3(768 / 64, 16384 / 64), 128, 0, stream>>>(
      gated, projwT, proj_b, (float*)d_out, 16384, 768, 768);
}
